// CausalSelfAttention_13322988552526
// MI455X (gfx1250) — compile-verified, matrix operand order checked by reading, not run on hardware
//
#include <hip/hip_runtime.h>
#include <hip/hip_bf16.h>

// ---------------------------------------------------------------------------
// CausalSelfAttention on gfx1250 (MI455X): bf16 WMMA pipeline, v3
//   B=4, T=2048, H=1024, NH=16, HD=64
// ---------------------------------------------------------------------------

#define B_ 4
#define T_ 2048
#define H_ 1024
#define NH_ 16
#define HD_ 64
#define MM_ (B_ * T_)   // 8192 rows in flattened [B*T, H]

typedef __attribute__((ext_vector_type(16))) __bf16 v16bf;
typedef __attribute__((ext_vector_type(8)))  __bf16 v8bf;
typedef __attribute__((ext_vector_type(8)))  float  v8f;
typedef __attribute__((ext_vector_type(4)))  float  v4f;
typedef __attribute__((ext_vector_type(4)))  unsigned short v4u16;

static __device__ __forceinline__ unsigned short f2bf(float f) {
    unsigned u = __builtin_bit_cast(unsigned, f);
    u += 0x7FFFu + ((u >> 16) & 1u);        // round-to-nearest-even
    return (unsigned short)(u >> 16);
}

static __device__ __forceinline__ v16bf cat16(v8bf lo, v8bf hi) {
    return __builtin_shufflevector(lo, hi, 0,1,2,3,4,5,6,7,8,9,10,11,12,13,14,15);
}

// load one 16x32 bf16 WMMA operand (A-layout == B-layout for row-major [16,K])
static __device__ __forceinline__ v16bf ldop(const unsigned short* row, int kc, int h8) {
    return cat16(*(const v8bf*)(row + kc + h8),
                 *(const v8bf*)(row + kc + 16 + h8));
}

// ---------------------------------------------------------------------------
// elementwise fp32 -> bf16 (vectorized x4; all sizes are multiples of 4)
// ---------------------------------------------------------------------------
__global__ void cvt_bf16_kernel(const float* __restrict__ in,
                                unsigned short* __restrict__ out, int n4) {
    int i = blockIdx.x * blockDim.x + threadIdx.x;
    int stride = gridDim.x * blockDim.x;
    const v4f* pin  = (const v4f*)in;
    v4u16*     pout = (v4u16*)out;
    for (; i < n4; i += stride) {
        v4f v = pin[i];
        v4u16 o;
        o[0] = f2bf(v[0]); o[1] = f2bf(v[1]); o[2] = f2bf(v[2]); o[3] = f2bf(v[3]);
        pout[i] = o;
    }
}

// ---------------------------------------------------------------------------
// C = A * W^T + bias ;  A:[MM,H] bf16 row-major, W:[H,H] bf16 row-major.
// One wave -> 32(M) x 64(N) tile. Ping-pong double buffering (no reg moves,
// no WMMA->VALU WAR hazards): MMA(bufA); load bufA; MMA(bufB); load bufB.
// Output scattered to head-major bf16:
//   transposed==0 : out[((b*NH+h)*T + t)*HD + d]   (Q, K)
//   transposed==1 : out[((b*NH+h)*HD + d)*T + t]   (V^T)
// grid: 1024 blocks x 128 threads = 4096 waves = (MM/32)*(H/64)
// ---------------------------------------------------------------------------
__global__ __launch_bounds__(128) void qkv_gemm_kernel(
    const unsigned short* __restrict__ A,
    const unsigned short* __restrict__ W,
    const float* __restrict__ bias,
    unsigned short* __restrict__ out,
    int transposed)
{
    const int lane = threadIdx.x & 31;
    const int widx = threadIdx.x >> 5;
    const int w    = blockIdx.x * 4 + widx;
    const int wn   = w & 15;        // H/64 = 16 N-tiles
    const int wm   = w >> 4;        // MM/32 = 256 M-tiles
    const int n    = lane & 15;
    const int hh   = lane >> 4;
    const int h8   = hh * 8;

    const unsigned short* arow0 = A + (size_t)(wm * 32 + n) * H_;
    const unsigned short* arow1 = arow0 + (size_t)16 * H_;
    const unsigned short* wrow[4];
    #pragma unroll
    for (int t = 0; t < 4; ++t)
        wrow[t] = W + (size_t)(wn * 64 + t * 16 + n) * H_;

    v8f acc[8];
    #pragma unroll
    for (int t = 0; t < 8; ++t) acc[t] = v8f{};

    auto loadCh = [&](int kc, v16bf& A0, v16bf& A1, v16bf (&BB)[4]) {
        A0 = ldop(arow0, kc, h8);
        A1 = ldop(arow1, kc, h8);
        #pragma unroll
        for (int t = 0; t < 4; ++t) BB[t] = ldop(wrow[t], kc, h8);
    };
    auto mmaCh = [&](const v16bf& A0, const v16bf& A1, const v16bf (&BB)[4]) {
        #pragma unroll
        for (int t = 0; t < 4; ++t) {
            acc[t]     = __builtin_amdgcn_wmma_f32_16x16x32_bf16(
                             false, A0, false, BB[t], (short)0, acc[t],     false, false);
            acc[4 + t] = __builtin_amdgcn_wmma_f32_16x16x32_bf16(
                             false, A1, false, BB[t], (short)0, acc[4 + t], false, false);
        }
    };

    v16bf a0A, a1A, bA[4];
    v16bf a0B, a1B, bB[4];
    loadCh(0,  a0A, a1A, bA);
    loadCh(32, a0B, a1B, bB);

    for (int kc = 64; kc < H_; kc += 64) {
        mmaCh(a0A, a1A, bA);            // compute chunk kc-64
        loadCh(kc, a0A, a1A, bA);       // refill in place (VMEM writes, no moves)
        mmaCh(a0B, a1B, bB);            // compute chunk kc-32
        loadCh(kc + 32, a0B, a1B, bB);
    }
    mmaCh(a0A, a1A, bA);                // chunk H_-64
    mmaCh(a0B, a1B, bB);                // chunk H_-32

    #pragma unroll
    for (int half = 0; half < 2; ++half) {
        #pragma unroll
        for (int t = 0; t < 4; ++t) {
            const int   j    = wn * 64 + t * 16 + n;   // output feature
            const float bv   = bias[j];
            const int   hidx = j >> 6;                 // / HD
            const int   d    = j & 63;
            #pragma unroll
            for (int r = 0; r < 8; ++r) {
                const int i    = wm * 32 + half * 16 + r + 8 * hh;  // row in [0, MM)
                const int bidx = i >> 11;                           // / T
                const int tseq = i & 2047;
                size_t o;
                if (transposed)
                    o = (((size_t)bidx * NH_ + hidx) * HD_ + d) * T_ + tseq;
                else
                    o = (((size_t)bidx * NH_ + hidx) * T_ + tseq) * HD_ + d;
                out[o] = f2bf(acc[half * 4 + t][r] + bv);
            }
        }
    }
}

// ---------------------------------------------------------------------------
// Flash attention (causal). One wave per 16-query tile, 32 keys per step.
// K tiles are register double-buffered (next-iter loads issue before the
// softmax); V loads issue right after the score WMMAs and overlap softmax.
//   qb,kb : [B,NH,T,HD] bf16   vtb : [B,NH,HD,T] bf16
//   out   : [B,NH,T,HD] fp32
// grid: 2048 blocks x 128 threads = 8192 waves = B*NH*(T/16)
// ---------------------------------------------------------------------------
__global__ __launch_bounds__(128) void attn_kernel(
    const unsigned short* __restrict__ qb,
    const unsigned short* __restrict__ kb,
    const unsigned short* __restrict__ vtb,
    const float* __restrict__ amask,
    float* __restrict__ out)
{
    __shared__ __align__(16) unsigned short Pl[4][16][32];

    const int lane = threadIdx.x & 31;
    const int widx = threadIdx.x >> 5;
    const int w    = blockIdx.x * 4 + widx;
    const int qt   = w & 127;        // T/16 = 128 q tiles
    const int bh   = w >> 7;         // b*NH + h
    const int b    = bh >> 4;
    const int q0   = qt * 16;

    const int n  = lane & 15;
    const int hh = lane >> 4;
    const int h8 = hh * 8;

    // Q tile 16x64 bf16 -> two A operands (HD split into K-chunks of 32)
    const unsigned short* qrow = qb + ((size_t)bh * T_ + q0 + n) * HD_;
    v16bf aq[2];
    #pragma unroll
    for (int c = 0; c < 2; ++c) aq[c] = ldop(qrow, c * 32, h8);

    const unsigned short* kbase = kb  + (size_t)bh * T_ * HD_;
    const unsigned short* vbase = vtb + (size_t)bh * HD_ * T_;
    const float*          mrow  = amask + (size_t)b * T_;

    float m_run[8], l_run[8];
    v8f acc[4];
    #pragma unroll
    for (int r = 0; r < 8; ++r) { m_run[r] = -3.0e38f; l_run[r] = 0.f; }
    #pragma unroll
    for (int t = 0; t < 4; ++t) acc[t] = v8f{};

    // bk[0..1] = key-tile 0/1 for HD chunk 0; bk[2..3] = same for HD chunk 1
    auto loadK = [&](int k0, v16bf (&bk)[4]) {
        #pragma unroll
        for (int c = 0; c < 2; ++c) {
            bk[c * 2 + 0] = ldop(kbase + (size_t)(k0 + n)      * HD_, c * 32, h8);
            bk[c * 2 + 1] = ldop(kbase + (size_t)(k0 + 16 + n) * HD_, c * 32, h8);
        }
    };

    const int nIter = (q0 + 47) >> 5;   // ceil((q0+16)/32)

    // one 32-key step: scores from `cur`, prefetch next K into `nxt`
    auto step = [&](int it, const v16bf (&cur)[4], v16bf (&nxt)[4], bool loadNext) {
        const int k0 = it * 32;

        // ---- scores S = Q * K^T (16 x 32), two 16x16 WMMA tiles ----
        v8f s0 = v8f{}, s1 = v8f{};
        s0 = __builtin_amdgcn_wmma_f32_16x16x32_bf16(
                 false, aq[0], false, cur[0], (short)0, s0, false, false);
        s1 = __builtin_amdgcn_wmma_f32_16x16x32_bf16(
                 false, aq[0], false, cur[1], (short)0, s1, false, false);
        s0 = __builtin_amdgcn_wmma_f32_16x16x32_bf16(
                 false, aq[1], false, cur[2], (short)0, s0, false, false);
        s1 = __builtin_amdgcn_wmma_f32_16x16x32_bf16(
                 false, aq[1], false, cur[3], (short)0, s1, false, false);

        // ---- issue V loads; they complete while the softmax runs ----
        v16bf bv[4];
        #pragma unroll
        for (int t = 0; t < 4; ++t)
            bv[t] = ldop(vbase + (size_t)(t * 16 + n) * T_, k0, h8);

        // ---- issue next iteration's K loads before the softmax chain ----
        if (loadNext) {
            loadK(k0 + 32, nxt);
            __builtin_prefetch(kbase + (size_t)(k0 + 64 + n) * HD_, 0, 1);
            __builtin_prefetch(kbase + (size_t)(k0 + 80 + n) * HD_, 0, 1);
        }

        const float am0 = mrow[k0 + n];
        const float am1 = mrow[k0 + 16 + n];

        // ---- causal mask + online softmax (rows span 16 lanes) ----
        float p0[8], p1[8], alpha[8];
        #pragma unroll
        for (int r = 0; r < 8; ++r) {
            const int q  = q0 + r + 8 * hh;      // global query index for slot r
            float sc0 = (k0 + n      <= q) ? s0[r] * 0.125f + am0 : -1.0e30f;
            float sc1 = (k0 + 16 + n <= q) ? s1[r] * 0.125f + am1 : -1.0e30f;
            float tm = fmaxf(sc0, sc1);
            tm = fmaxf(tm, __shfl_xor(tm, 1, 32));
            tm = fmaxf(tm, __shfl_xor(tm, 2, 32));
            tm = fmaxf(tm, __shfl_xor(tm, 4, 32));
            tm = fmaxf(tm, __shfl_xor(tm, 8, 32));
            const float mnew = fmaxf(m_run[r], tm);
            alpha[r] = __expf(m_run[r] - mnew);
            p0[r]    = __expf(sc0 - mnew);
            p1[r]    = __expf(sc1 - mnew);
            float rs = p0[r] + p1[r];
            rs += __shfl_xor(rs, 1, 32);
            rs += __shfl_xor(rs, 2, 32);
            rs += __shfl_xor(rs, 4, 32);
            rs += __shfl_xor(rs, 8, 32);
            l_run[r] = l_run[r] * alpha[r] + rs;
            m_run[r] = mnew;
        }
        #pragma unroll
        for (int t = 0; t < 4; ++t)
            #pragma unroll
            for (int r = 0; r < 8; ++r)
                acc[t][r] *= alpha[r];

        // ---- D-layout P -> A-layout via LDS round trip ----
        #pragma unroll
        for (int r = 0; r < 8; ++r) {
            Pl[widx][r + 8 * hh][n]      = f2bf(p0[r]);
            Pl[widx][r + 8 * hh][n + 16] = f2bf(p1[r]);
        }
        asm volatile("s_wait_dscnt 0x0" ::: "memory");
        const unsigned short* prow = &Pl[widx][n][0];
        v16bf ap = cat16(*(const v8bf*)(prow + h8), *(const v8bf*)(prow + 16 + h8));

        // ---- O += P * V  (4 HD tiles of 16) ----
        #pragma unroll
        for (int t = 0; t < 4; ++t)
            acc[t] = __builtin_amdgcn_wmma_f32_16x16x32_bf16(
                         false, ap, false, bv[t], (short)0, acc[t], false, false);
    };

    v16bf bkA[4], bkB[4];
    loadK(0, bkA);
    int it = 0;
    while (true) {
        const bool next1 = (it + 1 < nIter);
        step(it, bkA, bkB, next1);      // consumes A, refills B
        ++it;
        if (!next1) break;
        const bool next2 = (it + 1 < nIter);
        step(it, bkB, bkA, next2);      // consumes B, refills A
        ++it;
        if (!next2) break;
    }

    // ---- normalize and store fp32 [B,NH,T,HD] ----
    float inv[8];
    #pragma unroll
    for (int r = 0; r < 8; ++r) inv[r] = 1.0f / l_run[r];
    float* orow = out + ((size_t)bh * T_ + q0) * HD_;
    #pragma unroll
    for (int t = 0; t < 4; ++t)
        #pragma unroll
        for (int r = 0; r < 8; ++r)
            orow[(size_t)(r + 8 * hh) * HD_ + t * 16 + n] = acc[t][r] * inv[r];
}

// ---------------------------------------------------------------------------
extern "C" void kernel_launch(void* const* d_in, const int* in_sizes, int n_in,
                              void* d_out, int out_size, void* d_ws, size_t ws_size,
                              hipStream_t stream) {
    const float* hidden = (const float*)d_in[0];
    const float* amask  = (const float*)d_in[1];
    const float* Wq     = (const float*)d_in[2];
    const float* bq     = (const float*)d_in[3];
    const float* Wk     = (const float*)d_in[4];
    const float* bk     = (const float*)d_in[5];
    const float* Wv     = (const float*)d_in[6];
    const float* bv     = (const float*)d_in[7];
    float* out = (float*)d_out;

    char* ws = (char*)d_ws;
    size_t off = 0;
    auto grab = [&](size_t bytes) -> char* {
        char* p = ws + off;
        off += (bytes + 255) & ~(size_t)255;
        return p;
    };
    unsigned short* hb  = (unsigned short*)grab((size_t)MM_ * H_ * 2);
    unsigned short* wqb = (unsigned short*)grab((size_t)H_ * H_ * 2);
    unsigned short* wkb = (unsigned short*)grab((size_t)H_ * H_ * 2);
    unsigned short* wvb = (unsigned short*)grab((size_t)H_ * H_ * 2);
    unsigned short* qbf = (unsigned short*)grab((size_t)MM_ * H_ * 2);
    unsigned short* kbf = (unsigned short*)grab((size_t)MM_ * H_ * 2);
    unsigned short* vtb = (unsigned short*)grab((size_t)MM_ * H_ * 2);

    cvt_bf16_kernel<<<2048, 256, 0, stream>>>(hidden, hb, MM_ * H_ / 4);
    cvt_bf16_kernel<<<512,  256, 0, stream>>>(Wq, wqb, H_ * H_ / 4);
    cvt_bf16_kernel<<<512,  256, 0, stream>>>(Wk, wkb, H_ * H_ / 4);
    cvt_bf16_kernel<<<512,  256, 0, stream>>>(Wv, wvb, H_ * H_ / 4);

    qkv_gemm_kernel<<<1024, 128, 0, stream>>>(hb, wqb, bq, qbf, 0);
    qkv_gemm_kernel<<<1024, 128, 0, stream>>>(hb, wkb, bk, kbf, 0);
    qkv_gemm_kernel<<<1024, 128, 0, stream>>>(hb, wvb, bv, vtb, 1);

    attn_kernel<<<2048, 128, 0, stream>>>(qbf, kbf, vtb, amask, out);
}
